// Interaction_53644141527205
// MI455X (gfx1250) — compile-verified, matrix operand order checked by reading, not run on hardware
//
#include <hip/hip_runtime.h>
#include <math.h>

// ---------------- problem constants ----------------
#define BB   8
#define NUM  4
#define BN   (BB*NUM)   // 32 fused batches
#define LQ   128
#define LP   512
#define HH   512
#define H5   (5*HH)     // 2560

// LDS tile row stride (u16 elems): 32 payload + 8 pad = 80B rows (16B aligned).
#define SSTR 40

typedef __attribute__((ext_vector_type(16))) __bf16          v16bf;
typedef __attribute__((ext_vector_type(8)))  float           v8f;
typedef __attribute__((ext_vector_type(8)))  unsigned short  v8u16;
typedef __attribute__((ext_vector_type(16))) unsigned short  v16u16;
typedef __attribute__((ext_vector_type(4)))  unsigned int    u32x4;
typedef __attribute__((ext_vector_type(8)))  int             i32x8;
typedef __attribute__((ext_vector_type(4)))  int             i32x4;

// fp32 -> bf16, round to nearest even
__device__ __forceinline__ unsigned short f2bf(float f) {
    unsigned u = __float_as_uint(f);
    u += 0x7fffu + ((u >> 16) & 1u);
    return (unsigned short)(u >> 16);
}

__device__ __forceinline__ unsigned lds_off(const void* p) {
    // generic pointers to LDS carry the LDS byte offset in the low 32 bits
    return (unsigned)(size_t)p;
}

// ---------------- CDNA5 async global->LDS (per-lane, ASYNCcnt) ----------------
__device__ __forceinline__ void async_cp16(unsigned short* sh, const unsigned short* g) {
    asm volatile("global_load_async_to_lds_b128 %0, %1, off"
                 :: "v"(lds_off(sh)), "v"((unsigned long long)(size_t)g)
                 : "memory");
}
__device__ __forceinline__ void wait_async() {
    asm volatile("s_wait_asynccnt 0x0" ::: "memory");
}
__device__ __forceinline__ void wait_tensor() {
#if __has_builtin(__builtin_amdgcn_s_wait_tensorcnt)
    __builtin_amdgcn_s_wait_tensorcnt(0);
#else
    asm volatile("s_wait_tensorcnt 0x0" ::: "memory");
#endif
}

// ---------------- CDNA5 Tensor Data Mover: 128-row x 32-col bf16 tile ----------
// D#: 2B elements, tile_dim0=32 (contiguous k), tile_dim1=128 rows,
// stride0=ld; LDS padding: pad_interval=16 DWORDs (one 64B row),
// pad_amount=4 DWORDs (16B) -> reproduces the SSTR=40 padded layout.
__device__ __forceinline__ void tdm_load_tile_bf16(unsigned lds_addr,
                                                   const unsigned short* g,
                                                   unsigned ld_elems) {
    unsigned long long ga = (unsigned long long)(size_t)g;
    u32x4 g0;
    g0[0] = 1u;                                            // count=1 (valid), user mode
    g0[1] = lds_addr;                                      // lds_addr
    g0[2] = (unsigned)ga;                                  // global_addr[31:0]
    g0[3] = (unsigned)((ga >> 32) & 0x1FFFFFFu) | (2u << 30); // addr[56:32] | type=2
    i32x8 g1;
    g1[0] = (1 << 16)        // data_size = 2 bytes
          | (1 << 20)        // pad_enable
          | (3 << 22)        // pad_interval: 16 DWORDs
          | (3 << 25);       // pad_amount:   4 DWORDs
    g1[1] = (int)(32u << 16);   // tensor_dim0[15:0]=32 (in [31:16]); barrier addr=0
    g1[2] = (int)(128u << 16);  // tensor_dim0 hi=0 | tensor_dim1[15:0]=128
    g1[3] = (int)(32u << 16);   // tensor_dim1 hi=0 | tile_dim0=32
    g1[4] = 128;                // tile_dim1=128 | tile_dim2=0
    g1[5] = (int)ld_elems;      // tensor_dim0_stride[31:0]
    g1[6] = 0;
    g1[7] = 0;
#if __has_builtin(__builtin_amdgcn_tensor_load_to_lds)
    i32x4 z4 = (i32x4)0;
    i32x8 z8 = (i32x8)0;
    __builtin_amdgcn_tensor_load_to_lds(g0, g1, z4, z4, z8, 0);
#else
    asm volatile("tensor_load_to_lds %0, %1" :: "s"(g0), "s"(g1) : "memory");
#endif
}

// ---------------- LDS staging helpers (256 threads) ----------------
// Async copy of a 128-row x 32-col bf16 tile, row-major source (pure memcpy).
__device__ __forceinline__ void async_stage_row(unsigned short* sh,
                                                const unsigned short* __restrict__ src,
                                                int ld, int kbase) {
    const int t   = threadIdx.x;
    const int row = t >> 1;
    const int c0  = (t & 1) << 4;
    const unsigned short* g = src + (size_t)row * ld + kbase + c0;
    unsigned short* l = sh + row * SSTR + c0;
    async_cp16(l,     g);
    async_cp16(l + 8, g + 8);
}

// Transpose-on-store staging of a K x N (32 x 128) tile from a row-major
// K-major fp32 source: sh[n][kk] = bf16(src[(kbase+kk)*ld + nbase + n]).
__device__ __forceinline__ void stage_kn(unsigned short* sh, const float* __restrict__ src,
                                         int ld, int kbase, int nbase) {
    const int t   = threadIdx.x;
    const int n0  = (t & 31) << 2;
    const int kk0 = t >> 5;
#pragma unroll
    for (int kk = kk0; kk < 32; kk += 8) {
        const float4 v = *(const float4*)(src + (size_t)(kbase + kk) * ld + nbase + n0);
        sh[(n0 + 0) * SSTR + kk] = f2bf(v.x);
        sh[(n0 + 1) * SSTR + kk] = f2bf(v.y);
        sh[(n0 + 2) * SSTR + kk] = f2bf(v.z);
        sh[(n0 + 3) * SSTR + kk] = f2bf(v.w);
    }
}

// ---------------- WMMA fragment load + 16x16x32 bf16 MMA ----------------
__device__ __forceinline__ v16bf load_frag(const unsigned short* sh, int row, int hf) {
    const unsigned short* p = sh + row * SSTR + (hf << 3);
    v8u16 lo = *(const v8u16*)(p);
    v8u16 hi = *(const v8u16*)(p + 16);
    v16u16 t;
#pragma unroll
    for (int i = 0; i < 8; ++i) { t[i] = lo[i]; t[i + 8] = hi[i]; }
    return __builtin_bit_cast(v16bf, t);
}

// Software-pipelined: B fragment nt+1 is fetched from LDS before the WMMA
// consuming fragment nt issues, so only partial s_wait_dscnt is needed.
__device__ __forceinline__ void mma_tiles(const unsigned short* As, const unsigned short* Bs,
                                          v8f acc[8]) {
    const int lane = threadIdx.x & 31;
    const int wave = threadIdx.x >> 5;
    const int hf   = lane >> 4;
    const int r    = lane & 15;
    v16bf a    = load_frag(As, wave * 16 + r, hf);
    v16bf bcur = load_frag(Bs, r, hf);
#pragma unroll
    for (int nt = 0; nt < 8; ++nt) {
        v16bf bnxt;
        if (nt < 7) bnxt = load_frag(Bs, (nt + 1) * 16 + r, hf);
        acc[nt] = __builtin_amdgcn_wmma_f32_16x16x32_bf16(
            false, a, false, bcur, (short)0, acc[nt], false, false);
        if (nt < 7) bcur = bnxt;
    }
}

// Double-buffered GEMM inner loop: async A tiles + caller-supplied B staging.
template <typename FB>
__device__ __forceinline__ void gemm_db(const unsigned short* __restrict__ Ag, int lda, int K,
                                        FB&& stageB,
                                        unsigned short (&As)[2][128 * SSTR],
                                        unsigned short (&Bs)[2][128 * SSTR],
                                        v8f (&acc)[8]) {
    async_stage_row(As[0], Ag, lda, 0);
    stageB(Bs[0], 0);
    wait_async();
    __syncthreads();
    int cur = 0;
    for (int k = 0; k < K; k += 32) {
        const int nxt = cur ^ 1;
        if (k + 32 < K) {
            async_stage_row(As[nxt], Ag, lda, k + 32);
            stageB(Bs[nxt], k + 32);
        }
        mma_tiles(As[cur], Bs[cur], acc);
        wait_async();
        __syncthreads();
        cur = nxt;
    }
}

// ---------------- K-1: convert fp32 -> bf16 (optional per-column scale) -----
__global__ __launch_bounds__(256) void k_conv(const float* __restrict__ src,
                                              unsigned short* __restrict__ dst,
                                              const float* __restrict__ scale, int use_scale) {
    const size_t i = ((size_t)blockIdx.x * 256 + threadIdx.x) * 8;
    float4 a = *(const float4*)(src + i);
    float4 b = *(const float4*)(src + i + 4);
    if (use_scale) {
        const float* s = scale + (i & (HH - 1));   // HH is a power of two
        a.x *= s[0]; a.y *= s[1]; a.z *= s[2]; a.w *= s[3];
        b.x *= s[4]; b.y *= s[5]; b.z *= s[6]; b.w *= s[7];
    }
    v8u16 o;
    o[0] = f2bf(a.x); o[1] = f2bf(a.y); o[2] = f2bf(a.z); o[3] = f2bf(a.w);
    o[4] = f2bf(b.x); o[5] = f2bf(b.y); o[6] = f2bf(b.z); o[7] = f2bf(b.w);
    *(v8u16*)(dst + i) = o;
}

// ---------------- K0: bias dot products ----------------
__global__ __launch_bounds__(256) void k_bias(const float* __restrict__ Eq,
                                              const float* __restrict__ Ep,
                                              const float* __restrict__ w,
                                              float* __restrict__ aq,
                                              float* __restrict__ ap) {
    const int row  = blockIdx.x * 8 + (threadIdx.x >> 5);
    const int lane = threadIdx.x & 31;
    const float* src; const float* wv; float* dst;
    if (row < BN * LQ) { src = Eq + (size_t)row * HH;           wv = w;      dst = aq + row; }
    else               { int r2 = row - BN * LQ;
                         src = Ep + (size_t)r2 * HH;            wv = w + HH; dst = ap + r2; }
    float s = 0.f;
#pragma unroll
    for (int i = 0; i < HH / 32; ++i) s += src[lane + 32 * i] * wv[lane + 32 * i];
#pragma unroll
    for (int off = 16; off; off >>= 1) s += __shfl_xor(s, off, 32);
    if (lane == 0) *dst = s;
}

// ---------------- K1: U = (Ep.*wm) @ Eq^T + biases, masked ----------------
// A tile via TDM (wave 0 issues tensor_load_to_lds), B tile via async loads.
__global__ __launch_bounds__(256) void k_u(const unsigned short* __restrict__ EpW,
                                           const unsigned short* __restrict__ Eqb,
                                           const int*   __restrict__ mq,
                                           const int*   __restrict__ mp,
                                           const float* __restrict__ aq,
                                           const float* __restrict__ ap,
                                           float* __restrict__ U) {
    __shared__ __align__(16) unsigned short As[2][128 * SSTR];
    __shared__ __align__(16) unsigned short Bs[2][128 * SSTR];
    const int p0 = blockIdx.x * 128;
    const int bn = blockIdx.y;
    const unsigned short* Ag = EpW + (size_t)(bn * LP + p0) * HH;
    const unsigned short* Bg = Eqb + (size_t)bn * LQ * HH;
    const bool w0 = threadIdx.x < 32;   // wave-uniform
    v8f acc[8];
#pragma unroll
    for (int i = 0; i < 8; ++i) acc[i] = (v8f)0.f;

    if (w0) tdm_load_tile_bf16(lds_off(As[0]), Ag, HH);
    async_stage_row(Bs[0], Bg, HH, 0);
    wait_async();
    if (w0) wait_tensor();
    __syncthreads();
    int cur = 0;
    for (int k = 0; k < HH; k += 32) {
        const int nxt = cur ^ 1;
        if (k + 32 < HH) {
            if (w0) tdm_load_tile_bf16(lds_off(As[nxt]), Ag + (k + 32), HH);
            async_stage_row(Bs[nxt], Bg, HH, k + 32);
        }
        mma_tiles(As[cur], Bs[cur], acc);
        wait_async();
        if (w0) wait_tensor();
        __syncthreads();
        cur = nxt;
    }

    const int lane = threadIdx.x & 31, wave = threadIdx.x >> 5;
    const int hf = lane >> 4, r = lane & 15;
    float apv[8]; int mpv[8];
#pragma unroll
    for (int v = 0; v < 8; ++v) {
        const int p = p0 + wave * 16 + v + 8 * hf;
        apv[v] = ap[bn * LP + p];
        mpv[v] = mp[bn * LP + p];
    }
#pragma unroll
    for (int nt = 0; nt < 8; ++nt) {
        const int q     = nt * 16 + r;
        const float aqv = aq[bn * LQ + q];
        const int   mqv = mq[bn * LQ + q];
#pragma unroll
        for (int v = 0; v < 8; ++v) {
            const int p = p0 + wave * 16 + v + 8 * hf;
            float val = acc[nt][v] + apv[v] + aqv;
            U[(size_t)(bn * LP + p) * LQ + q] = (mqv && mpv[v]) ? val : -3.402823466e38f;
        }
    }
}

// ---------------- K2: row softmax (over q) -> Ap_bf (row-major p x q) -------
__global__ __launch_bounds__(256) void k_softmax_row(const float* __restrict__ U,
                                                     const int* __restrict__ mq,
                                                     const int* __restrict__ mp,
                                                     unsigned short* __restrict__ Apb) {
    const int row  = blockIdx.x * 8 + (threadIdx.x >> 5);   // bn*LP + p
    const int lane = threadIdx.x & 31;
    const int bn   = row / LP;
    const float* u = U + (size_t)row * LQ;
    float x[4]; float mx = -INFINITY;
#pragma unroll
    for (int j = 0; j < 4; ++j) { x[j] = u[lane + 32 * j]; mx = fmaxf(mx, x[j]); }
#pragma unroll
    for (int off = 16; off; off >>= 1) mx = fmaxf(mx, __shfl_xor(mx, off, 32));
    float s = 0.f;
#pragma unroll
    for (int j = 0; j < 4; ++j) { x[j] = __expf(x[j] - mx); s += x[j]; }
#pragma unroll
    for (int off = 16; off; off >>= 1) s += __shfl_xor(s, off, 32);
    const float inv = 1.f / s;
    const int mpv = mp[row];
#pragma unroll
    for (int j = 0; j < 4; ++j) {
        const int q = lane + 32 * j;
        Apb[(size_t)row * LQ + q] =
            f2bf((mpv && mq[bn * LQ + q]) ? x[j] * inv : 0.f);
    }
}

// ---------------- K3: column softmax (over p) -> BpT_bf (q x p, transposed) -
__global__ __launch_bounds__(256) void k_softmax_col(const float* __restrict__ U,
                                                     const int* __restrict__ mq,
                                                     const int* __restrict__ mp,
                                                     unsigned short* __restrict__ BpTb) {
    const int col  = blockIdx.x * 8 + (threadIdx.x >> 5);   // bn*LQ + q
    const int lane = threadIdx.x & 31;
    const int bn = col / LQ;
    const int q  = col & (LQ - 1);
    const float* u = U + (size_t)bn * LP * LQ + q;
    float x[16]; float mx = -INFINITY;
#pragma unroll
    for (int j = 0; j < 16; ++j) { x[j] = u[(size_t)(lane + 32 * j) * LQ]; mx = fmaxf(mx, x[j]); }
#pragma unroll
    for (int off = 16; off; off >>= 1) mx = fmaxf(mx, __shfl_xor(mx, off, 32));
    float s = 0.f;
#pragma unroll
    for (int j = 0; j < 16; ++j) { x[j] = __expf(x[j] - mx); s += x[j]; }
#pragma unroll
    for (int off = 16; off; off >>= 1) s += __shfl_xor(s, off, 32);
    const float inv = 1.f / s;
    const int mqv = mq[col];
#pragma unroll
    for (int j = 0; j < 16; ++j) {
        const int p = lane + 32 * j;
        BpTb[(size_t)col * LP + p] =              // contiguous across lanes
            f2bf((mqv && mp[bn * LP + p]) ? x[j] * inv : 0.f);
    }
}

// ---------------- K4: A1 = A_p @ E_q ----------------
__global__ __launch_bounds__(256) void k_a1(const unsigned short* __restrict__ Apb,
                                            const float* __restrict__ Eq,
                                            float* __restrict__ A1) {
    __shared__ __align__(16) unsigned short As[2][128 * SSTR];
    __shared__ __align__(16) unsigned short Bs[2][128 * SSTR];
    const int n0 = blockIdx.x * 128, p0 = blockIdx.y * 128, bn = blockIdx.z;
    v8f acc[8];
#pragma unroll
    for (int i = 0; i < 8; ++i) acc[i] = (v8f)0.f;
    const float* Bsrc = Eq + (size_t)bn * LQ * HH;
    gemm_db(Apb + (size_t)(bn * LP + p0) * LQ, LQ, LQ,
            [&](unsigned short* sh, int k) { stage_kn(sh, Bsrc, HH, k, n0); },
            As, Bs, acc);
    const int lane = threadIdx.x & 31, wave = threadIdx.x >> 5;
    const int hf = lane >> 4, r = lane & 15;
#pragma unroll
    for (int nt = 0; nt < 8; ++nt)
#pragma unroll
        for (int v = 0; v < 8; ++v) {
            const int p = p0 + wave * 16 + v + 8 * hf;
            const int h = n0 + nt * 16 + r;
            A1[(size_t)(bn * LP + p) * HH + h] = acc[nt][v];
        }
}

// ---------------- K5: B1 = B_p^T @ E_p ----------------
__global__ __launch_bounds__(256) void k_b1(const unsigned short* __restrict__ BpTb,
                                            const float* __restrict__ Ep,
                                            float* __restrict__ B1) {
    __shared__ __align__(16) unsigned short As[2][128 * SSTR];
    __shared__ __align__(16) unsigned short Bs[2][128 * SSTR];
    const int n0 = blockIdx.x * 128, bn = blockIdx.y;
    v8f acc[8];
#pragma unroll
    for (int i = 0; i < 8; ++i) acc[i] = (v8f)0.f;
    const float* Bsrc = Ep + (size_t)bn * LP * HH;
    gemm_db(BpTb + (size_t)bn * LQ * LP, LP, LP,
            [&](unsigned short* sh, int k) { stage_kn(sh, Bsrc, HH, k, n0); },
            As, Bs, acc);
    const int lane = threadIdx.x & 31, wave = threadIdx.x >> 5;
    const int hf = lane >> 4, r = lane & 15;
#pragma unroll
    for (int nt = 0; nt < 8; ++nt)
#pragma unroll
        for (int v = 0; v < 8; ++v) {
            const int q = wave * 16 + v + 8 * hf;
            const int h = n0 + nt * 16 + r;
            B1[(size_t)(bn * LQ + q) * HH + h] = acc[nt][v];
        }
}

// ---------------- K6: A2 = A_p @ B1, fused G_q_p epilogue ----------------
__global__ __launch_bounds__(256) void k_a2_out(const unsigned short* __restrict__ Apb,
                                                const float* __restrict__ B1,
                                                const float* __restrict__ A1,
                                                const float* __restrict__ Ep,
                                                const int*   __restrict__ mp,
                                                float* __restrict__ out2) {
    __shared__ __align__(16) unsigned short As[2][128 * SSTR];
    __shared__ __align__(16) unsigned short Bs[2][128 * SSTR];
    const int n0 = blockIdx.x * 128, p0 = blockIdx.y * 128, bn = blockIdx.z;
    v8f acc[8];
#pragma unroll
    for (int i = 0; i < 8; ++i) acc[i] = (v8f)0.f;
    const float* Bsrc = B1 + (size_t)bn * LQ * HH;
    gemm_db(Apb + (size_t)(bn * LP + p0) * LQ, LQ, LQ,
            [&](unsigned short* sh, int k) { stage_kn(sh, Bsrc, HH, k, n0); },
            As, Bs, acc);
    const int lane = threadIdx.x & 31, wave = threadIdx.x >> 5;
    const int hf = lane >> 4, r = lane & 15;
#pragma unroll
    for (int nt = 0; nt < 8; ++nt)
#pragma unroll
        for (int v = 0; v < 8; ++v) {
            const int p = p0 + wave * 16 + v + 8 * hf;
            const int h = n0 + nt * 16 + r;
            const size_t idx = (size_t)(bn * LP + p) * HH + h;
            const float mm = mp[bn * LP + p] ? 1.f : 0.f;
            const float ep = Ep[idx], a1 = A1[idx], a2 = acc[nt][v];
            float* o = out2 + (size_t)(bn * LP + p) * H5 + h;
            o[0 * HH] = ep * mm;
            o[1 * HH] = a1 * mm;
            o[2 * HH] = a2 * mm;
            o[3 * HH] = ep * a1 * mm;
            o[4 * HH] = ep * a2 * mm;
        }
}

// ---------------- K7: B2 = B_p^T @ A1, fused G_p_q epilogue ----------------
__global__ __launch_bounds__(256) void k_b2_out(const unsigned short* __restrict__ BpTb,
                                                const float* __restrict__ A1,
                                                const float* __restrict__ B1,
                                                const float* __restrict__ Eq,
                                                const int*   __restrict__ mq,
                                                float* __restrict__ out1) {
    __shared__ __align__(16) unsigned short As[2][128 * SSTR];
    __shared__ __align__(16) unsigned short Bs[2][128 * SSTR];
    const int n0 = blockIdx.x * 128, bn = blockIdx.y;
    v8f acc[8];
#pragma unroll
    for (int i = 0; i < 8; ++i) acc[i] = (v8f)0.f;
    const float* Bsrc = A1 + (size_t)bn * LP * HH;
    gemm_db(BpTb + (size_t)bn * LQ * LP, LP, LP,
            [&](unsigned short* sh, int k) { stage_kn(sh, Bsrc, HH, k, n0); },
            As, Bs, acc);
    const int lane = threadIdx.x & 31, wave = threadIdx.x >> 5;
    const int hf = lane >> 4, r = lane & 15;
#pragma unroll
    for (int nt = 0; nt < 8; ++nt)
#pragma unroll
        for (int v = 0; v < 8; ++v) {
            const int q = wave * 16 + v + 8 * hf;
            const int h = n0 + nt * 16 + r;
            const size_t idx = (size_t)(bn * LQ + q) * HH + h;
            const float mm = mq[bn * LQ + q] ? 1.f : 0.f;
            const float eq = Eq[idx], b1 = B1[idx], b2 = acc[nt][v];
            float* o = out1 + (size_t)(bn * LQ + q) * H5 + h;
            o[0 * HH] = eq * mm;
            o[1 * HH] = b1 * mm;
            o[2 * HH] = b2 * mm;
            o[3 * HH] = eq * b1 * mm;
            o[4 * HH] = eq * b2 * mm;
        }
}

// ---------------- launch ----------------
extern "C" void kernel_launch(void* const* d_in, const int* in_sizes, int n_in,
                              void* d_out, int out_size, void* d_ws, size_t ws_size,
                              hipStream_t stream) {
    const float* Eq = (const float*)d_in[0];   // (B,NUM,LQ,H)
    const float* Ep = (const float*)d_in[1];   // (B,NUM,LP,H)
    const int*   mq = (const int*)d_in[2];     // (B,NUM,LQ)
    const int*   mp = (const int*)d_in[3];     // (B,NUM,LP)
    const float* w  = (const float*)d_in[4];   // (3H,)

    float* ws = (float*)d_ws;
    float* aq = ws;                            // BN*LQ
    float* ap = aq + BN * LQ;                  // BN*LP
    float* U  = ap + BN * LP;                  // BN*LP*LQ
    float* A1 = U  + (size_t)BN * LP * LQ;     // BN*LP*H
    float* B1 = A1 + (size_t)BN * LP * HH;     // BN*LQ*H
    unsigned short* EpWb = (unsigned short*)(B1 + (size_t)BN * LQ * HH);
    unsigned short* Eqb  = EpWb + (size_t)BN * LP * HH;   // BN*LQ*H
    unsigned short* Apb  = Eqb  + (size_t)BN * LQ * HH;   // BN*LP*LQ
    unsigned short* BpTb = Apb  + (size_t)BN * LP * LQ;   // BN*LQ*LP

    float* out1 = (float*)d_out;                          // G_p_q: BN*LQ*5H
    float* out2 = out1 + (size_t)BN * LQ * H5;            // G_q_p: BN*LP*5H

    const float* wm = w + 2 * HH;

    k_conv<<<dim3((BN * LP * HH) / 2048), 256, 0, stream>>>(Ep, EpWb, wm, 1);
    k_conv<<<dim3((BN * LQ * HH) / 2048), 256, 0, stream>>>(Eq, Eqb, nullptr, 0);
    k_bias<<<dim3((BN * (LQ + LP)) / 8), 256, 0, stream>>>(Eq, Ep, w, aq, ap);
    k_u<<<dim3(LP / 128, BN), 256, 0, stream>>>(EpWb, Eqb, mq, mp, aq, ap, U);
    k_softmax_col<<<dim3(BN * LQ / 8), 256, 0, stream>>>(U, mq, mp, BpTb);
    k_softmax_row<<<dim3(BN * LP / 8), 256, 0, stream>>>(U, mq, mp, Apb);
    k_a1<<<dim3(HH / 128, LP / 128, BN), 256, 0, stream>>>(Apb, Eq, A1);
    k_b1<<<dim3(HH / 128, BN), 256, 0, stream>>>(BpTb, Ep, B1);
    k_a2_out<<<dim3(HH / 128, LP / 128, BN), 256, 0, stream>>>(Apb, B1, A1, Ep, mp, out2);
    k_b2_out<<<dim3(HH / 128, BN), 256, 0, stream>>>(BpTb, A1, B1, Eq, mq, out1);
}
